// VectorQuantizer_42966852829620
// MI455X (gfx1250) — matrix operand order repaired and checked by reading, not compiled, not run on hardware
//
#include <hip/hip_runtime.h>
#include <math.h>

typedef float    v2f  __attribute__((ext_vector_type(2)));
typedef float    v8f  __attribute__((ext_vector_type(8)));
typedef _Float16 v8h  __attribute__((ext_vector_type(8)));
typedef _Float16 v16h __attribute__((ext_vector_type(16)));

#define D        64
#define K        1024
#define NTOK     262144
#define CHUNK    256      // codebook rows staged in LDS per pass
#define BLK_TOK  128      // tokens per block (8 waves x 16 tokens)
#define THREADS  256
#define NBLOCKS  (NTOK / BLK_TOK)   // 2048
#define XPAD     68       // f32 row pad: 272B, 16B aligned, distinct banks across rows
#define CPAD     88       // f16 row pad: 176B, 16B aligned, 44-bank stride -> conflict-free

// ---------------- prep: codeword norms + hi/lo f16 split + zero histogram ----------------
__global__ __launch_bounds__(256) void vq_prep(const float* __restrict__ cb,
                                               float* __restrict__ cnorm,
                                               int* __restrict__ counts,
                                               _Float16* __restrict__ cbh,
                                               _Float16* __restrict__ cbl) {
    int k = blockIdx.x * blockDim.x + threadIdx.x;
    if (k < K) {
        float s = 0.f;
#pragma unroll
        for (int i = 0; i < D; ++i) {
            float x = cb[(size_t)k * D + i];
            s += x * x;
            _Float16 h = (_Float16)x;
            cbh[(size_t)k * D + i] = h;
            cbl[(size_t)k * D + i] = (_Float16)(x - (float)h);
        }
        cnorm[k]  = s;      // exact fp32 ||c||^2
        counts[k] = 0;
    }
}

__device__ __forceinline__ v16h ld16h(const _Float16* p) {
    v8h a = *(const v8h*)p;
    v8h b = *(const v8h*)(p + 16);
    return __builtin_shufflevector(a, b, 0,1,2,3,4,5,6,7,8,9,10,11,12,13,14,15);
}

// ---------------- main: split-f16 WMMA distance argmin + gather + partial loss ----------------
__global__ __launch_bounds__(THREADS) void vq_main(const float* __restrict__ X,
                                                   const float* __restrict__ CB,
                                                   const _Float16* __restrict__ cbh,
                                                   const _Float16* __restrict__ cbl,
                                                   const float* __restrict__ cnorm,
                                                   int* __restrict__ counts,
                                                   float* __restrict__ partials,
                                                   float* __restrict__ out) {
    __shared__ float    sX[BLK_TOK][XPAD];
    __shared__ _Float16 sCBh[CHUNK][CPAD];
    __shared__ _Float16 sCBl[CHUNK][CPAD];
    __shared__ int      sIdx[BLK_TOK];
    __shared__ float    sRed[THREADS];

    const int    tid     = threadIdx.x;
    const int    wave    = tid >> 5;
    const int    lane    = tid & 31;
    const int    col     = lane & 15;                // C/D column slot, also A row slot
    const int    ka      = (lane & 16) ? 8 : 0;      // A-matrix K offset for upper half-wave
    const int    kb      = (lane & 16) ? 8 : 0;      // B-matrix K offset for upper half-wave
    const size_t tokBase = (size_t)blockIdx.x * BLK_TOK;

    // ---- stage X tile (fp32) into LDS ----
    {
        const float4* g = (const float4*)(X + tokBase * D);
        for (int e = tid; e < BLK_TOK * (D / 4); e += THREADS) {
            float4 v = g[e];
            int r = e / (D / 4), c = (e % (D / 4)) * 4;
            *(float4*)&sX[r][c] = v;   // 16B aligned: row pad 68 floats
        }
    }
    __syncthreads();

    // ---- per-wave split A fragments: 2 K-chunks of 16x32, hi and lo planes ----
    v16h ah[2], al[2];
    {
        const int m = wave * 16 + col;
#pragma unroll
        for (int q = 0; q < 2; ++q) {
            const int Q = q * 32;
#pragma unroll
            for (int t = 0; t < 8; ++t) {
                float x0 = sX[m][Q + ka + t];          // K = Q+ka .. Q+ka+7      -> halves 0..7
                float x1 = sX[m][Q + 16 + ka + t];     // K = Q+16+ka .. +7       -> halves 8..15
                _Float16 h0 = (_Float16)x0;
                _Float16 h1 = (_Float16)x1;
                ah[q][t]     = h0;
                ah[q][8 + t] = h1;
                al[q][t]     = (_Float16)(x0 - (float)h0);
                al[q][8 + t] = (_Float16)(x1 - (float)h1);
            }
        }
    }

    float best[8];
    int   bidx[8];
#pragma unroll
    for (int j = 0; j < 8; ++j) { best[j] = 3.402823466e+38f; bidx[j] = 0; }

    for (int cc = 0; cc < K / CHUNK; ++cc) {
        __syncthreads();   // protect previous chunk reads
        {
            // copy pre-split codebook chunk (hi/lo) into padded LDS rows; 8 halves per op
            const v8h* gh = (const v8h*)(cbh + (size_t)cc * CHUNK * D);
            const v8h* gl = (const v8h*)(cbl + (size_t)cc * CHUNK * D);
            for (int e = tid; e < CHUNK * (D / 8); e += THREADS) {
                int r = e / (D / 8), c = (e % (D / 8)) * 8;
                *(v8h*)&sCBh[r][c] = gh[e];
                *(v8h*)&sCBl[r][c] = gl[e];
            }
        }
        __syncthreads();

        for (int blk = 0; blk < CHUNK / 16; ++blk) {
            const int r  = blk * 16 + col;          // codebook row in LDS chunk (this lane's column)
            const int cw = cc * CHUNK + r;          // global codeword id
            // B fragments: K halves kb..kb+7 and 16+kb..16+kb+7 per lane, per K-chunk, hi & lo
            v16h bh0 = ld16h(&sCBh[r][kb]);
            v16h bh1 = ld16h(&sCBh[r][32 + kb]);
            v16h bl0 = ld16h(&sCBl[r][kb]);
            v16h bl1 = ld16h(&sCBl[r][32 + kb]);

            v8f acc = {0.f, 0.f, 0.f, 0.f, 0.f, 0.f, 0.f, 0.f};
            // x.c  ~=  hi.hi + hi.lo + lo.hi   (fp32 accumulation; lo.lo ~ 2^-22 dropped)
            acc = __builtin_amdgcn_wmma_f32_16x16x32_f16(false, ah[0], false, bl0, (short)0, acc, false, false);
            acc = __builtin_amdgcn_wmma_f32_16x16x32_f16(false, ah[1], false, bl1, (short)0, acc, false, false);
            acc = __builtin_amdgcn_wmma_f32_16x16x32_f16(false, al[0], false, bh0, (short)0, acc, false, false);
            acc = __builtin_amdgcn_wmma_f32_16x16x32_f16(false, al[1], false, bh1, (short)0, acc, false, false);
            acc = __builtin_amdgcn_wmma_f32_16x16x32_f16(false, ah[0], false, bh0, (short)0, acc, false, false);
            acc = __builtin_amdgcn_wmma_f32_16x16x32_f16(false, ah[1], false, bh1, (short)0, acc, false, false);

            const float cn = cnorm[cw];
#pragma unroll
            for (int j = 0; j < 8; ++j) {
                float score = fmaf(-2.0f, acc[j], cn);  // ||c||^2 - 2 x.c  (||x||^2 const per row)
                if (score < best[j]) { best[j] = score; bidx[j] = cw; }
            }
        }
    }

    // ---- argmin across the 16 lanes of each half-wave (rows j and j+8) ----
#pragma unroll
    for (int j = 0; j < 8; ++j) {
        float b  = best[j];
        int   i0 = bidx[j];
#pragma unroll
        for (int off = 8; off >= 1; off >>= 1) {
            float ob = __shfl_xor(b, off, 16);
            int   oi = __shfl_xor(i0, off, 16);
            if (ob < b || (ob == b && oi < i0)) { b = ob; i0 = oi; }  // first-min tie-break
        }
        if (col == 0) {
            const int m = wave * 16 + j + ((lane >> 4) * 8);
            sIdx[m] = i0;
        }
    }
    __syncthreads();

    // ---- histogram (deterministic integer atomics) ----
    if (tid < BLK_TOK) atomicAdd(&counts[sIdx[tid]], 1);

    // ---- gather exact fp32 codewords, write output, accumulate squared error ----
    float lsum = 0.f;
    for (int e = tid; e < BLK_TOK * (D / 4); e += THREADS) {
        int r = e / (D / 4), c = (e % (D / 4)) * 4;
        const float4 q = *(const float4*)(CB + (size_t)sIdx[r] * D + c);
        float dx = q.x - sX[r][c + 0];
        float dy = q.y - sX[r][c + 1];
        float dz = q.z - sX[r][c + 2];
        float dw = q.w - sX[r][c + 3];
        lsum += dx * dx + dy * dy + dz * dz + dw * dw;
        *(float4*)(out + (tokBase + r) * D + c) = q;
    }
    sRed[tid] = lsum;
    __syncthreads();
    for (int s = THREADS / 2; s > 0; s >>= 1) {
        if (tid < s) sRed[tid] += sRed[tid + s];
        __syncthreads();
    }
    if (tid == 0) partials[blockIdx.x] = sRed[0];
}

// ---------------- finalize: loss + perplexity (fixed-order reductions) ----------------
__global__ __launch_bounds__(1024) void vq_final(const int* __restrict__ counts,
                                                 const float* __restrict__ partials,
                                                 float* __restrict__ out_scalars) {
    __shared__ float rl[1024];
    __shared__ float rh[1024];
    const int t = threadIdx.x;
    rl[t] = partials[t] + partials[t + 1024];      // NBLOCKS == 2048
    float p = (float)counts[t] / (float)NTOK;
    rh[t] = p * logf(p + 1e-5f);
    __syncthreads();
    for (int s = 512; s > 0; s >>= 1) {
        if (t < s) { rl[t] += rl[t + s]; rh[t] += rh[t + s]; }
        __syncthreads();
    }
    if (t == 0) {
        out_scalars[0] = 1.25f * rl[0] / (float)(NTOK * D);  // codebook + 0.25*commitment
        out_scalars[1] = expf(-rh[0]);                       // perplexity
    }
}

extern "C" void kernel_launch(void* const* d_in, const int* in_sizes, int n_in,
                              void* d_out, int out_size, void* d_ws, size_t ws_size,
                              hipStream_t stream) {
    const float* X  = (const float*)d_in[0];   // [NTOK, 64] fp32
    const float* CB = (const float*)d_in[1];   // [1024, 64] fp32
    float* out = (float*)d_out;                // [NTOK*64] quantized, then loss, perplexity

    // workspace layout:
    //   cnorm    : 1024 f32              @ 0
    //   counts   : 1024 i32              @ 4096
    //   partials : 2048 f32              @ 8192
    //   cbh      : 1024*64 f16 (128 KB)  @ 16384
    //   cbl      : 1024*64 f16 (128 KB)  @ 147456
    float*    cnorm    = (float*)d_ws;
    int*      counts   = (int*)((char*)d_ws + 4096);
    float*    partials = (float*)((char*)d_ws + 8192);
    _Float16* cbh      = (_Float16*)((char*)d_ws + 16384);
    _Float16* cbl      = (_Float16*)((char*)d_ws + 16384 + (size_t)K * D * sizeof(_Float16));

    vq_prep<<<(K + 255) / 256, 256, 0, stream>>>(CB, cnorm, counts, cbh, cbl);
    vq_main<<<NBLOCKS, THREADS, 0, stream>>>(X, CB, cbh, cbl, cnorm, counts, partials, out);
    vq_final<<<1, 1024, 0, stream>>>(counts, partials, out + (size_t)NTOK * D);
}
